// NeuralNDCGLoss_89988154786182
// MI455X (gfx1250) — 8-bit matrix operand order repaired and checked by reading, compiled, not run on hardware
//
#include <hip/hip_runtime.h>

#define NN    512
#define NB    256
#define FEPS  1e-10f
#define ITERS 50

typedef int   v16i __attribute__((ext_vector_type(16)));
typedef float v8f  __attribute__((ext_vector_type(8)));

__device__ __forceinline__ unsigned char enc8(float x) {
    unsigned u  = __float_as_uint(x);
    unsigned sg = (u >> 24) & 0x80u;
    float ax = fabsf(x);
    if (!(ax == ax)) return 0;
    if (ax >= 448.0f) return (unsigned char)(sg | 0x7Eu);
    if (ax < 0.015625f) {
        int m = (int)(ax * 512.0f + 0.5f);
        if (m >= 8) return (unsigned char)(sg | 0x08u);
        return (unsigned char)(sg | (unsigned)m);
    }
    unsigned ua   = u & 0x7FFFFFFFu;
    unsigned mant = ua & 0x7FFFFFu;
    int e8 = (int)(ua >> 23) - 127;
    unsigned m3 = (mant + 0x80000u) >> 20;
    int E = e8 + 7;
    if (m3 == 8u) { m3 = 0u; ++E; }
    return (unsigned char)(sg | ((unsigned)E << 3) | m3);
}

__device__ __forceinline__ float dec8(unsigned char b) {
    unsigned sg = (b >> 7) & 1u;
    int E = (b >> 3) & 0xF;
    int m = b & 7;
    float r = (E == 0) ? (float)m * 1.953125e-3f
                       : __uint_as_float(((unsigned)(E - 7 + 127) << 23) | ((unsigned)m << 20));
    return sg ? -r : r;
}

__device__ __forceinline__ void matvec_fp8(const unsigned char* __restrict__ mat,
                                           const unsigned* __restrict__ vhi,
                                           const unsigned* __restrict__ vlo,
                                           float* __restrict__ out) {
    const int  lane  = threadIdx.x & 31;
    const int  wave  = threadIdx.x >> 5;
    const int  lmod  = lane & 15;
    const int  lhi8  = (lane >> 4) << 3;
    const bool isB   = (lane == 0) || (lane == 1) || (lane == 16) || (lane == 17);
    const int  kaddB = (lane >= 16) ? 16 : 0;

    for (int g = wave; g < 16; g += 8) {
        const int M0a = g << 4;
        const int M0b = (g + 16) << 4;
        v8f acc0 = {0.f, 0.f, 0.f, 0.f, 0.f, 0.f, 0.f, 0.f};
        v8f acc1 = {0.f, 0.f, 0.f, 0.f, 0.f, 0.f, 0.f, 0.f};
        const unsigned* row0 = (const unsigned*)(mat + (((size_t)M0a + (size_t)lmod) << 9));
        const unsigned* row1 = (const unsigned*)(mat + (((size_t)M0b + (size_t)lmod) << 9));
        for (int kb = 0; kb < 4; ++kb) {
            const int K0 = kb << 7;
            if (kb < 3) {
                __builtin_prefetch((const void*)(row0 + ((K0 + 128) >> 2)), 0, 3);
                __builtin_prefetch((const void*)(row1 + ((K0 + 128) >> 2)), 0, 3);
            }
            v16i A0, A1;
#pragma unroll
            for (int d = 0; d < 16; ++d) {
                int dd   = d & 7;
                int koff = ((d >> 2) << 5) + ((d & 3) << 2) + (lhi8 << 1);
                A0[d] = (int)row0[(K0 + koff) >> 2];
                A1[d] = (int)row1[(K0 + koff) >> 2];
            }
            v16i Bv;
#pragma unroll
            for (int d = 0; d < 16; ++d) Bv[d] = 0;
            if (isB) {
                const unsigned* src = (lane & 1) ? vlo : vhi;
                const uint4* s4 = (const uint4*)(src + ((K0 + kaddB) >> 2));
                uint4 q0 = s4[0];
                uint4 q1 = s4[2];
                uint4 q2 = s4[4];
                uint4 q3 = s4[6];
                Bv[0]  = (int)q0.x; Bv[1]  = (int)q0.y; Bv[2]  = (int)q0.z; Bv[3]  = (int)q0.w;
                Bv[4]  = (int)q1.x; Bv[5]  = (int)q1.y; Bv[6]  = (int)q1.z; Bv[7]  = (int)q1.w;
                Bv[8]  = (int)q2.x; Bv[9]  = (int)q2.y; Bv[10] = (int)q2.z; Bv[11] = (int)q2.w;
                Bv[12] = (int)q3.x; Bv[13] = (int)q3.y; Bv[14] = (int)q3.z; Bv[15] = (int)q3.w;
            }
            acc0 = __builtin_amdgcn_wmma_f32_16x16x128_fp8_fp8(A0, Bv, (short)0, acc0, false, false);
            acc1 = __builtin_amdgcn_wmma_f32_16x16x128_fp8_fp8(A1, Bv, (short)0, acc1, false, false);
        }
        if (isB) {
            int Mb = (lane >= 16) ? 8 : 0;
#pragma unroll
            for (int v = 0; v < 8; ++v) {
                atomicAdd(&out[M0a + Mb + v], acc0[v]);
                atomicAdd(&out[M0b + Mb + v], acc1[v]);
            }
        }
    }
}

__global__ void ndcg_zero_acc(float* gacc) { gacc[0] = 0.f; gacc[1] = 0.f; }

__global__ void ndcg_finalize(const float* gacc, float* out) {
    out[0] = -gacc[0] / gacc[1];
}

__global__ void __launch_bounds__(256)
neural_ndcg_main(const float* __restrict__ y_pred, const float* __restrict__ y_true,
                 unsigned char* __restrict__ Eall, unsigned char* __restrict__ ETall,
                 float* __restrict__ gacc) {
    __shared__ float s_s[NN], s_B[NN], s_g[NN], s_yt[NN];
    __shared__ float s_r[NN], s_c[NN], s_acc[NN], s_cg[NN];
    __shared__ __align__(16) unsigned s_vhi[NN / 4];
    __shared__ __align__(16) unsigned s_vlo[NN / 4];
    __shared__ float s_tab[256];
    __shared__ int s_hist[16], s_pref[16];
    __shared__ float s_idcg, s_red;
    __shared__ unsigned s_max;

    const int b    = blockIdx.x;
    const int tid  = threadIdx.x;
    const int lane = tid & 31;
    const int wave = tid >> 5;
    unsigned char* Eb  = Eall  + ((size_t)b << 18);
    unsigned char* ETb = ETall + ((size_t)b << 18);

    for (int j = tid; j < NN; j += 256) {
        s_s[j]  = y_pred[(size_t)b * NN + j];
        float yt = y_true[(size_t)b * NN + j];
        s_yt[j] = yt;
        s_g[j]  = exp2f(yt) - 1.0f;
        s_c[j]  = 1.0f;
    }
    s_tab[tid] = dec8((unsigned char)tid);
    if (tid == 0) { s_idcg = 0.f; s_red = 0.f; }
    if (tid < 16) s_hist[tid] = 0;
    __syncthreads();

    for (int j = tid; j < NN; j += 256) {
        float sj = s_s[j], acc = 0.f;
        for (int k = 0; k < NN; ++k) acc += fabsf(sj - s_s[k]);
        s_B[j] = acc;
    }
    for (int j = tid; j < NN; j += 256) {
        int v = (int)(s_yt[j] + 0.5f);
        v = v < 0 ? 0 : (v > 15 ? 15 : v);
        atomicAdd(&s_hist[v], 1);
    }
    __syncthreads();
    if (tid == 0) {
        int pos = 0;
        for (int v = 15; v >= 0; --v) { s_pref[v] = pos; pos += s_hist[v]; }
    }
    __syncthreads();
    {
        float ip = 0.f;
        for (int p = tid; p < NN; p += 256) {
            int vp = 0;
            for (int v = 15; v >= 0; --v)
                if (p >= s_pref[v] && p < s_pref[v] + s_hist[v]) { vp = v; break; }
            ip += (exp2f((float)vp) - 1.0f) / __log2f((float)(p + 2));
        }
        atomicAdd(&s_idcg, ip);
    }

    for (int i = wave * 64; i < wave * 64 + 64; ++i) {
        float scal = (float)(511 - 2 * i);
        float pm[16];
        float mx = -3.4e38f;
#pragma unroll
        for (int jj = 0; jj < 16; ++jj) {
            int j = lane + (jj << 5);
            float p = scal * s_s[j] - s_B[j];
            pm[jj] = p;
            mx = fmaxf(mx, p);
        }
#pragma unroll
        for (int o = 16; o > 0; o >>= 1) mx = fmaxf(mx, __shfl_xor(mx, o, 32));
        float zs = 0.f;
#pragma unroll
        for (int jj = 0; jj < 16; ++jj) {
            int j = lane + (jj << 5);
            float e = __expf(pm[jj] - mx);
            zs += e;
            unsigned char q = enc8(e);
            Eb [(size_t)i * NN + j] = q;
            ETb[(size_t)j * NN + i] = q;
        }
#pragma unroll
        for (int o = 16; o > 0; o >>= 1) zs += __shfl_xor(zs, o, 32);
        if (lane == 0) s_r[i] = 1.0f / zs;
    }
    __threadfence();
    __syncthreads();

    for (int it = 0; it < ITERS; ++it) {
        if (tid == 0) s_max = 0u;
        __syncthreads();
        {
            unsigned mloc = 0u;
            for (int i = tid; i < NN; i += 256) {
                s_acc[i] = 0.f;
                unsigned u = __float_as_uint(s_r[i]);
                mloc = mloc > u ? mloc : u;
            }
            atomicMax(&s_max, mloc);
        }
        __syncthreads();
        float rmax = __uint_as_float(s_max);
        float rinv = 1.0f / rmax;
        for (int i = tid; i < NN; i += 256) {
            float q = s_r[i] * rinv;
            unsigned char h = enc8(q);
            ((unsigned char*)s_vhi)[i] = h;
            ((unsigned char*)s_vlo)[i] = enc8(q - dec8(h));
        }
        __syncthreads();
        matvec_fp8(ETb, s_vhi, s_vlo, s_acc);
        __syncthreads();
        for (int j = tid; j < NN; j += 256) {
            float w = s_acc[j] * rmax;
            float c = s_c[j];
            s_c[j] = c / fmaxf(c * w, FEPS);
        }
        if (tid == 0) s_max = 0u;
        __syncthreads();
        {
            unsigned mloc = 0u;
            for (int j = tid; j < NN; j += 256) {
                s_acc[j] = 0.f;
                unsigned u = __float_as_uint(s_c[j]);
                mloc = mloc > u ? mloc : u;
            }
            atomicMax(&s_max, mloc);
        }
        __syncthreads();
        float cmax = __uint_as_float(s_max);
        float cinv = 1.0f / cmax;
        for (int j = tid; j < NN; j += 256) {
            float q = s_c[j] * cinv;
            unsigned char h = enc8(q);
            ((unsigned char*)s_vhi)[j] = h;
            ((unsigned char*)s_vlo)[j] = enc8(q - dec8(h));
        }
        __syncthreads();
        matvec_fp8(Eb, s_vhi, s_vlo, s_acc);
        __syncthreads();
        for (int i = tid; i < NN; i += 256) {
            float v = s_acc[i] * cmax;
            float r = s_r[i];
            s_r[i] = r / fmaxf(r * v, FEPS);
        }
        __syncthreads();
    }

    for (int j = tid; j < NN; j += 256) s_cg[j] = s_c[j] * s_g[j];
    __syncthreads();
    float part = 0.f;
    for (int i = tid; i < NN; i += 256) {
        const unsigned* rp = (const unsigned*)(Eb + (size_t)i * NN);
        float acc = 0.f;
        for (int jq = 0; jq < NN / 4; ++jq) {
            unsigned q = rp[jq];
            int j4 = jq << 2;
            acc += s_tab[q & 255u]          * s_cg[j4]
                 + s_tab[(q >> 8)  & 255u]  * s_cg[j4 + 1]
                 + s_tab[(q >> 16) & 255u]  * s_cg[j4 + 2]
                 + s_tab[(q >> 24)]         * s_cg[j4 + 3];
        }
        float gt = s_r[i] * acc;
        part += gt / __log2f((float)(i + 2));
    }
    atomicAdd(&s_red, part);
    __syncthreads();
    if (tid == 0) {
        float idcg = s_idcg;
        if (idcg != 0.0f) {
            atomicAdd(&gacc[0], s_red / (idcg + FEPS));
            atomicAdd(&gacc[1], 1.0f);
        }
    }
}

extern "C" void kernel_launch(void* const* d_in, const int* in_sizes, int n_in,
                              void* d_out, int out_size, void* d_ws, size_t ws_size,
                              hipStream_t stream) {
    (void)in_sizes; (void)n_in; (void)out_size; (void)ws_size;
    const float* y_pred = (const float*)d_in[0];
    const float* y_true = (const float*)d_in[1];
    float* gacc = (float*)d_ws;
    unsigned char* E  = (unsigned char*)d_ws + 256;
    unsigned char* ET = E + ((size_t)NB << 18);
    ndcg_zero_acc<<<1, 1, 0, stream>>>(gacc);
    neural_ndcg_main<<<NB, 256, 0, stream>>>(y_pred, y_true, E, ET, gacc);
    ndcg_finalize<<<1, 1, 0, stream>>>(gacc, (float*)d_out);
}
